// CascadeTree_18382460027616
// MI455X (gfx1250) — matrix operand order repaired and checked by reading, not compiled, not run on hardware
//
#include <hip/hip_runtime.h>

typedef __attribute__((ext_vector_type(16))) _Float16 v16h;
typedef __attribute__((ext_vector_type(8)))  float    v8f;

#define NLVL 6
#define K_SMIN 32.0f

struct Params {
  const float* pt;
  const float* grid[NLVL];
  const float* w1[NLVL];
  const float* b1[NLVL];
  const float* w2[NLVL];
  const float* b2[NLVL];
  float* out;
};

__device__ __forceinline__ float smin_pair(float a, float b) {
  // iquilezles exp-smin with the reference's bad-range guard
  float r  = exp2f(-K_SMIN * a) + exp2f(-K_SMIN * b);
  float ar = fabsf(r);
  bool  bad = (ar < 1e-10f) || (ar > 1e10f);
  float mn  = fminf(a, b);
  float sv  = -log2f(bad ? 1.0f : r) * (1.0f / K_SMIN);
  return bad ? mn : sv;
}

__global__ __launch_bounds__(256) void cascade_wmma_kernel(Params P) {
  // w1 for all levels, f16, transposed to [n(64)][k(32)] so B-fragments are
  // contiguous b32 DS loads.  24 KB.
  __shared__ _Float16 sW1[NLVL][64][32];
  // per-wave 16-point x 32-feat f16 tile (A-matrix staging).  8 KB.
  __shared__ _Float16 sFeat[8][16][32];

  const int tid = threadIdx.x;

  // ---- block-wide preload: w1 (f32 global) -> f16 transposed LDS ----
  for (int i = tid; i < NLVL * 64 * 32; i += 256) {
    int lvl = i >> 11;            // /2048
    int rem = i & 2047;
    int n   = rem >> 5;           // 0..63
    int k   = rem & 31;           // 0..31
    sW1[lvl][n][k] = (_Float16)P.w1[lvl][k * 64 + n];
  }
  __syncthreads();

  const int wave = tid >> 5;
  const int lane = tid & 31;
  const int p    = lane & 15;     // point within wave tile / fragment row-col id
  const int g    = lane >> 4;     // half-group (feature half 16g..16g+15)
  const int base_pt = blockIdx.x * 128 + wave * 16;

  // each lane loads its point's coords (redundant x2 across halves; cached)
  const float* pp = P.pt + 3 * (base_pt + p);
  float alpha[3];
  alpha[0] = (pp[0] + 1.0f) * 0.5f;
  alpha[1] = (pp[1] + 1.0f) * 0.5f;
  alpha[2] = (pp[2] + 1.0f) * 0.5f;

  float acc[8];
#pragma unroll
  for (int r = 0; r < 8; ++r) acc[r] = 0.0f;

  for (int lvl = 0; lvl < NLVL; ++lvl) {
    const int   res = 4 << lvl;
    const float rm1 = (float)(res - 1);

    int   ix[3];
    float fr[3];
#pragma unroll
    for (int d = 0; d < 3; ++d) {
      float a  = alpha[d] * rm1;
      float fl = floorf(a);
      int   ii = (int)fl;
      ii = ii < 0 ? 0 : (ii > res - 1 ? res - 1 : ii);
      ix[d] = ii;
      fr[d] = a - fl;
    }

    // ---- trilinear gather: this lane accumulates feats [16g..16g+15] ----
    float f[16];
#pragma unroll
    for (int j = 0; j < 16; ++j) f[j] = 0.0f;

    const float* gptr = P.grid[lvl];
#pragma unroll
    for (int c = 0; c < 8; ++c) {
      const int cx = (c >> 2) & 1, cy = (c >> 1) & 1, cz = c & 1;
      float w = (cx ? fr[0] : 1.0f - fr[0]) *
                (cy ? fr[1] : 1.0f - fr[1]) *
                (cz ? fr[2] : 1.0f - fr[2]);
      int x = ix[0] + cx; x = x > res - 1 ? res - 1 : x;
      int y = ix[1] + cy; y = y > res - 1 ? res - 1 : y;
      int z = ix[2] + cz; z = z > res - 1 ? res - 1 : z;
      const float4* b4 =
          (const float4*)(gptr + (((x * res + y) * res + z) << 5) + (g << 4));
#pragma unroll
      for (int q = 0; q < 4; ++q) {
        float4 v = b4[q];
        f[4 * q + 0] += w * v.x;
        f[4 * q + 1] += w * v.y;
        f[4 * q + 2] += w * v.z;
        f[4 * q + 3] += w * v.w;
      }
    }

    // ---- pack f32 -> f16 pairs into the wave's LDS A-tile ----
    unsigned int* ft = (unsigned int*)&sFeat[wave][p][g * 16];
#pragma unroll
    for (int j = 0; j < 8; ++j) {
      union { _Float16 h[2]; unsigned int u; } pk;
      pk.h[0] = (_Float16)f[2 * j];
      pk.h[1] = (_Float16)f[2 * j + 1];
      ft[j] = pk.u;
    }
    asm volatile("s_wait_dscnt 0" ::: "memory");  // intra-wave LDS RAW fence

    // ---- A fragment: 16x32 f16, contiguous k order (not the ISA table order) ----
    // VGPR v : K = 16g + 2v,+1 (contiguous, matches B)
    union { v16h v; unsigned int u[8]; } A;
    const unsigned int* lf = (const unsigned int*)&sFeat[wave][p][0];
#pragma unroll
    for (int vv = 0; vv < 4; ++vv) {
      A.u[vv]     = lf[8 * g + vv];
      A.u[4 + vv] = lf[8 * g + 4 + vv];
    }

    // ---- 32->64 layer: 4 N-tiles of v_wmma_f32_16x16x32_f16 ----
    v8f D[4];
#pragma unroll
    for (int t = 0; t < 4; ++t) {
      // B fragment 32x16: lane col n=p, K = 16g + j (contiguous in sW1[n][k])
      union { v16h v; unsigned int u[8]; } B;
      const unsigned int* wb =
          (const unsigned int*)&sW1[lvl][t * 16 + p][g * 16];
#pragma unroll
      for (int vv = 0; vv < 8; ++vv) B.u[vv] = wb[vv];
      v8f Cz = {};
      D[t] = __builtin_amdgcn_wmma_f32_16x16x32_f16(
          false, A.v, false, B.v, (short)0, Cz, false, false);
    }

    // ---- 64->1 layer on the D fragment (lane n=p, VGPR r -> row 8g+r) ----
    float partial[8];
#pragma unroll
    for (int r = 0; r < 8; ++r) partial[r] = 0.0f;
#pragma unroll
    for (int t = 0; t < 4; ++t) {
      float b1v = P.b1[lvl][t * 16 + p];
      float w2v = P.w2[lvl][t * 16 + p];
#pragma unroll
      for (int r = 0; r < 8; ++r) {
        float h = D[t][r] + b1v;
        h = h > 0.0f ? h : 0.0f;
        partial[r] += h * w2v;
      }
    }
    float b2v = P.b2[lvl][0];
#pragma unroll
    for (int r = 0; r < 8; ++r) {
      float s = partial[r];
      s += __shfl_xor(s, 1, 32);   // xor<16 keeps the two halves separate
      s += __shfl_xor(s, 2, 32);
      s += __shfl_xor(s, 4, 32);
      s += __shfl_xor(s, 8, 32);
      float lv = s + b2v;
      acc[r] = (lvl == 0) ? lv : smin_pair(acc[r], lv);
    }

    // warm next level's grid line for this point (speculative)
    if (lvl + 1 < NLVL) {
      const int r2 = res * 2;
      __builtin_prefetch(
          P.grid[lvl + 1] +
              (((ix[0] * 2) * r2 + ix[1] * 2) * r2 + ix[2] * 2) * 32,
          0, 1);
    }
  }

  // lane with col p==r in half-group g owns output point m = 8g + r
  float outv = acc[0];
#pragma unroll
  for (int r = 1; r < 8; ++r)
    if (p == r) outv = acc[r];
  if (p < 8) P.out[base_pt + 8 * g + p] = outv;
}

extern "C" void kernel_launch(void* const* d_in, const int* in_sizes, int n_in,
                              void* d_out, int out_size, void* d_ws, size_t ws_size,
                              hipStream_t stream) {
  Params P;
  P.pt = (const float*)d_in[0];
  for (int j = 0; j < NLVL; ++j) {
    P.grid[j] = (const float*)d_in[1 + 5 * j];
    P.w1[j]   = (const float*)d_in[2 + 5 * j];
    P.b1[j]   = (const float*)d_in[3 + 5 * j];
    P.w2[j]   = (const float*)d_in[4 + 5 * j];
    P.b2[j]   = (const float*)d_in[5 + 5 * j];
  }
  P.out = (float*)d_out;

  const int npts   = in_sizes[0] / 3;      // 524288
  const int blocks = npts / 128;           // 128 points per block (8 waves x 16)
  hipLaunchKernelGGL(cascade_wmma_kernel, dim3(blocks), dim3(256), 0, stream, P);
}
